// AGCRN_66417374265411
// MI455X (gfx1250) — compile-verified, matrix operand order checked by reading, not run on hardware
//
#include <hip/hip_runtime.h>
#include <hip/hip_bf16.h>
#include <math.h>

// ---------------- problem constants ----------------
#define N_NODES 1024
#define BATCH   16
#define HID     64
#define P_STEPS 24
#define SEQ_T   (N_NODES * BATCH * HID)   // elements per timestep in a sequence buffer
#define TILE_E  512                       // elements per swizzled 16x32 bf16 operand tile

typedef __attribute__((ext_vector_type(16))) __bf16 v16bf;
typedef __attribute__((ext_vector_type(8)))  float  v8f;

union Frag { v16bf v; uint4 q[2]; unsigned short s[16]; };

__device__ __forceinline__ unsigned short f2bf(float f) {
    unsigned u = __float_as_uint(f);
    u += 0x7fffu + ((u >> 16) & 1u);          // round-to-nearest-even
    return (unsigned short)(u >> 16);
}
__device__ __forceinline__ float bf2f(unsigned short s) {
    return __uint_as_float(((unsigned)s) << 16);
}

// Load a fragment (32 bytes, contiguous per lane) from a swizzled tile.
__device__ __forceinline__ v16bf ld_frag(const unsigned short* p) {
    Frag f;
    f.q[0] = *(const uint4*)(p);
    f.q[1] = *(const uint4*)(p + 8);
    return f.v;
}

// Swizzled tile layout (ISA 7.12.2, 16-bit 16x32 fragment):
//   tile[lane][j], lane = half*16 + (row-or-col), element j holds K = koff16(j,half)
//   half(K) = (K>>3)&1 ; j(K) = (K&7) + ((K&16)?8:0)
__device__ __forceinline__ void k_to_hj(int kk, int& hf, int& j) {
    hf = (kk >> 3) & 1;
    j  = (kk & 7) + ((kk & 16) ? 8 : 0);
}

// ---------------- adjacency: A = softmax(relu(E E^T)) -> swizzled bf16 A-operand tiles ----------------
// Aswz layout: [ntile(64)][mchunk(32)][TILE_E]
__global__ __launch_bounds__(128) void adj_kernel(const float* __restrict__ emb,
                                                  unsigned short* __restrict__ Aswz) {
    __shared__ float red[128];
    const int n = blockIdx.x, tid = threadIdx.x;
    float en[16];
#pragma unroll
    for (int e = 0; e < 16; ++e) en[e] = emb[n * 16 + e];
    float vals[8];
    float lmax = -1e30f;
#pragma unroll
    for (int i = 0; i < 8; ++i) {
        const int m = tid + i * 128;
        float s = 0.f;
#pragma unroll
        for (int e = 0; e < 16; ++e) s += en[e] * emb[m * 16 + e];
        s = fmaxf(s, 0.f);
        vals[i] = s;
        lmax = fmaxf(lmax, s);
    }
    red[tid] = lmax; __syncthreads();
    for (int st = 64; st > 0; st >>= 1) { if (tid < st) red[tid] = fmaxf(red[tid], red[tid + st]); __syncthreads(); }
    const float mx = red[0]; __syncthreads();
    float lsum = 0.f;
#pragma unroll
    for (int i = 0; i < 8; ++i) { vals[i] = expf(vals[i] - mx); lsum += vals[i]; }
    red[tid] = lsum; __syncthreads();
    for (int st = 64; st > 0; st >>= 1) { if (tid < st) red[tid] += red[tid + st]; __syncthreads(); }
    const float inv = 1.f / red[0];
    const int ntile = n >> 4, lmr = n & 15;
#pragma unroll
    for (int i = 0; i < 8; ++i) {
        const int m = tid + i * 128;
        int hf, j; k_to_hj(m & 31, hf, j);
        Aswz[((size_t)ntile * 32 + (m >> 5)) * TILE_E + (hf * 16 + lmr) * 16 + j] = f2bf(vals[i] * inv);
    }
}

// ---------------- patch embedding + positional encoding -> seq[t][(n*16+b)*64+h] bf16 ----------------
__global__ __launch_bounds__(256) void patch_kernel(const float* __restrict__ src,
                                                    const float* __restrict__ vew,
                                                    unsigned short* __restrict__ seq) {
    __shared__ float w[12 * 64];
    for (int i = threadIdx.x; i < 768; i += 256) w[i] = vew[i];
    __syncthreads();
    const size_t gid = (size_t)blockIdx.x * 256 + threadIdx.x;   // 24*1024*16
    const int b = (int)(gid & 15);
    const int n = (int)((gid >> 4) & 1023);
    const int p = (int)(gid >> 14);
    float xv[12];
#pragma unroll
    for (int t = 0; t < 12; ++t)
        xv[t] = src[((size_t)b * 288 + p * 12 + t) * N_NODES + n];
    unsigned short* o = seq + (size_t)p * SEQ_T + ((size_t)n * 16 + b) * HID;
#pragma unroll 4
    for (int h = 0; h < 64; ++h) {
        float acc = 0.f;
#pragma unroll
        for (int t = 0; t < 12; ++t) acc += xv[t] * w[t * 64 + h];
        const float arg = (float)p * expf(-(float)((h >> 1) * 2) * 0.14391157f); // ln(1e4)/64
        acc += (h & 1) ? cosf(arg) : sinf(arg);
        o[h] = f2bf(acc);
    }
}

// ---------------- per-node weight combine -> swizzled B-operand tiles ----------------
// Wout layout: [n][chunk(8)][ctile(OW/16)][TILE_E]
__global__ __launch_bounds__(256) void wcombine_kernel(const float* __restrict__ emb,
                                                       const float* __restrict__ gw,
                                                       unsigned short* __restrict__ Wout,
                                                       int OW) {
    const size_t gid = (size_t)blockIdx.x * 256 + threadIdx.x;   // N*256*(OW/4)
    const int ow4 = OW >> 2;
    const int o4 = (int)(gid % ow4);
    const int kc = (int)((gid / ow4) % 256);
    const int n  = (int)(gid / ((size_t)ow4 * 256));
    float a[4];
    a[0] = a[1] = a[2] = a[3] = 0.f;
#pragma unroll
    for (int e = 0; e < 16; ++e) {
        const float we = emb[n * 16 + e];
        const float* g = gw + ((size_t)e * 256 + kc) * OW + o4 * 4;
        a[0] += we * g[0]; a[1] += we * g[1]; a[2] += we * g[2]; a[3] += we * g[3];
    }
    const int nct = OW >> 4;
    int hf, j; k_to_hj(kc & 31, hf, j);
    unsigned short* base = Wout + (size_t)n * 8 * nct * TILE_E
                         + ((size_t)(kc >> 5) * nct) * TILE_E;
#pragma unroll
    for (int q = 0; q < 4; ++q) {
        const int o = o4 * 4 + q;
        base[(size_t)(o >> 4) * TILE_E + (hf * 16 + (o & 15)) * 16 + j] = f2bf(a[q]);
    }
}

// ---------------- per-node bias combine (f32) ----------------
__global__ __launch_bounds__(256) void bcombine_kernel(const float* __restrict__ emb,
                                                       const float* __restrict__ gb,
                                                       float* __restrict__ bout,
                                                       int OW) {
    const size_t gid = (size_t)blockIdx.x * 256 + threadIdx.x;   // N*OW
    const int o = (int)(gid % OW);
    const int n = (int)(gid / OW);
    float s = 0.f;
#pragma unroll
    for (int e = 0; e < 16; ++e) s += emb[n * 16 + e] * gb[e * OW + o];
    bout[gid] = s;
}

// Swizzled index for activations used as adjacency-GEMM B operand:
// Xswz layout: [b][mchunk(32)][ctile(8)][TILE_E], K dim = node m
__device__ __forceinline__ size_t act_swz(int b, int m, int c) {
    int hf, j; k_to_hj(m & 31, hf, j);
    return (((size_t)b * 32 + (m >> 5)) * 8 + (c >> 4)) * TILE_E + (hf * 16 + (c & 15)) * 16 + j;
}

// ---------------- pack inp = [x_t | state]: row-major bf16 + swizzled bf16 ----------------
__global__ __launch_bounds__(256) void pack_kernel(const unsigned short* __restrict__ seqIn,
                                                   const float* __restrict__ state,
                                                   unsigned short* __restrict__ inp,
                                                   unsigned short* __restrict__ inp_swz) {
    const size_t gid = (size_t)blockIdx.x * 256 + threadIdx.x;   // N*16*128
    const int c = (int)(gid & 127);
    const size_t r = gid >> 7;
    const unsigned short v = (c < 64) ? seqIn[r * 64 + c] : f2bf(state[r * 64 + (c - 64)]);
    inp[gid] = v;
    inp_swz[act_swz((int)(r & 15), (int)(r >> 4), c)] = v;
}

// ---------------- adjacency WMMA GEMM: Y[b][n,c] = sum_m A[n,m]*X[b][m,c] ----------------
// A from swizzled tiles, X from swizzled tiles, Y row-major bf16.
__global__ __launch_bounds__(128) void gemm_adj_kernel(const unsigned short* __restrict__ Aswz,
                                                       const unsigned short* __restrict__ Xswz,
                                                       unsigned short* __restrict__ Yout) {
    const int b     = blockIdx.y;
    const int ntile = blockIdx.x;
    const int tid   = threadIdx.x;
    const int wave  = tid >> 5, lane = tid & 31;
    const int half  = lane >> 4, lm = lane & 15;
    const int ct0   = wave * 2;
    v8f acc0 = {}, acc1 = {};
    const unsigned short* Ab = Aswz + (size_t)ntile * 32 * TILE_E + lane * 16;
    const unsigned short* Xb = Xswz + (size_t)b * 32 * 8 * TILE_E + lane * 16;
#pragma unroll 4
    for (int ch = 0; ch < 32; ++ch) {
        const v16bf af = ld_frag(Ab + (size_t)ch * TILE_E);
        const unsigned short* xt = Xb + ((size_t)ch * 8 + ct0) * TILE_E;
        const v16bf b0 = ld_frag(xt);
        const v16bf b1 = ld_frag(xt + TILE_E);
        acc0 = __builtin_amdgcn_wmma_f32_16x16x32_bf16(false, af, false, b0, (short)0, acc0, false, false);
        acc1 = __builtin_amdgcn_wmma_f32_16x16x32_bf16(false, af, false, b1, (short)0, acc1, false, false);
    }
#pragma unroll
    for (int v = 0; v < 8; ++v) {
        const int nrow = ntile * 16 + v + 8 * half;
        unsigned short* o = Yout + ((size_t)nrow * 16 + b) * 128 + ct0 * 16;
        o[lm]      = f2bf(acc0[v]);
        o[16 + lm] = f2bf(acc1[v]);
    }
}

// ---------------- gate: zr = sigmoid([inp|AX] @ Wg_n + bg_n); emit cand (both layouts), rbuf ----------------
__global__ __launch_bounds__(128) void gate_kernel(const unsigned short* __restrict__ inp,
                                                   const unsigned short* __restrict__ ax,
                                                   const unsigned short* __restrict__ Wg,
                                                   const float* __restrict__ bg,
                                                   const float* __restrict__ state,
                                                   unsigned short* __restrict__ cand,
                                                   unsigned short* __restrict__ cand_swz,
                                                   float* __restrict__ rbuf) {
    const int n    = blockIdx.x;
    const int tid  = threadIdx.x;
    const int wave = tid >> 5, lane = tid & 31;
    const int half = lane >> 4, lm = lane & 15;
    const int ct0  = wave * 2;
    const unsigned short* Wn = Wg + (size_t)n * 8 * 8 * TILE_E + lane * 16;
    v8f acc0 = {}, acc1 = {};
#pragma unroll
    for (int ch = 0; ch < 8; ++ch) {
        if (ch + 1 < 8)
            __builtin_prefetch((const void*)(Wn + (size_t)(ch + 1) * 8 * TILE_E), 0, 1);
        // A operand: activation row (K contiguous), rows = batches
        const unsigned short* Xs = (ch < 4) ? inp : ax;       // K = [inp(128) | AX(128)]
        const unsigned short* xr = Xs + ((size_t)n * 16 + lm) * 128 + ((ch * 32) & 127) + half * 8;
        Frag af;
        af.q[0] = *(const uint4*)(xr);
        af.q[1] = *(const uint4*)(xr + 16);
        const unsigned short* wt = Wn + ((size_t)ch * 8 + ct0) * TILE_E;
        const v16bf b0 = ld_frag(wt);
        const v16bf b1 = ld_frag(wt + TILE_E);
        acc0 = __builtin_amdgcn_wmma_f32_16x16x32_bf16(false, af.v, false, b0, (short)0, acc0, false, false);
        acc1 = __builtin_amdgcn_wmma_f32_16x16x32_bf16(false, af.v, false, b1, (short)0, acc1, false, false);
    }
    const float* bgn = bg + (size_t)n * 128;
#pragma unroll
    for (int v = 0; v < 8; ++v) {
        const int bb = v + 8 * half;                    // batch index (D-matrix row)
        const size_t r = (size_t)n * 16 + bb;
#pragma unroll
        for (int tt = 0; tt < 2; ++tt) {
            const int o = (ct0 + tt) * 16 + lm;
            const float val = (tt ? acc1[v] : acc0[v]) + bgn[o];
            const float s = 1.f / (1.f + expf(-val));
            if (o < 64) {                               // z gate -> candidate second half
                const unsigned short zv = f2bf(s * state[r * 64 + o]);
                cand[r * 128 + 64 + o] = zv;
                cand_swz[act_swz(bb, n, 64 + o)] = zv;
            } else {                                    // r gate -> rbuf; copy x_t into cand first half
                const int o2 = o - 64;
                rbuf[r * 64 + o2] = s;
                const unsigned short xv = inp[r * 128 + o2];
                cand[r * 128 + o2] = xv;
                cand_swz[act_swz(bb, n, o2)] = xv;
            }
        }
    }
}

// ---------------- update: hc = tanh([cand|AX2] @ Wu_n + bu_n); h = r*state + (1-r)*hc ----------------
__global__ __launch_bounds__(128) void update_kernel(const unsigned short* __restrict__ cand,
                                                     const unsigned short* __restrict__ ax,
                                                     const unsigned short* __restrict__ Wu,
                                                     const float* __restrict__ bu,
                                                     const float* __restrict__ rbuf,
                                                     float* __restrict__ state,
                                                     unsigned short* __restrict__ seqOut) {
    const int n    = blockIdx.x;
    const int tid  = threadIdx.x;
    const int wave = tid >> 5, lane = tid & 31;
    const int half = lane >> 4, lm = lane & 15;
    const int ct   = wave;                              // 4 waves x 1 ctile = 64 outputs
    const unsigned short* Wn = Wu + (size_t)n * 8 * 4 * TILE_E + lane * 16;
    v8f acc = {};
#pragma unroll
    for (int ch = 0; ch < 8; ++ch) {
        if (ch + 1 < 8)
            __builtin_prefetch((const void*)(Wn + (size_t)(ch + 1) * 4 * TILE_E), 0, 1);
        const unsigned short* Xs = (ch < 4) ? cand : ax;
        const unsigned short* xr = Xs + ((size_t)n * 16 + lm) * 128 + ((ch * 32) & 127) + half * 8;
        Frag af;
        af.q[0] = *(const uint4*)(xr);
        af.q[1] = *(const uint4*)(xr + 16);
        const v16bf bfr = ld_frag(Wn + ((size_t)ch * 4 + ct) * TILE_E);
        acc = __builtin_amdgcn_wmma_f32_16x16x32_bf16(false, af.v, false, bfr, (short)0, acc, false, false);
    }
    const float* bun = bu + (size_t)n * 64;
    const int o = ct * 16 + lm;
#pragma unroll
    for (int v = 0; v < 8; ++v) {
        const int bb = v + 8 * half;
        const size_t r = (size_t)n * 16 + bb;
        const float hc = tanhf(acc[v] + bun[o]);
        const float rr = rbuf[r * 64 + o];
        const float st = state[r * 64 + o];
        const float h = rr * st + (1.f - rr) * hc;
        state[r * 64 + o] = h;
        seqOut[r * 64 + o] = f2bf(h);
    }
}

// ---------------- horizon head ----------------
__global__ __launch_bounds__(256) void final_kernel(const unsigned short* __restrict__ last,
                                                    const float* __restrict__ ecw,
                                                    const float* __restrict__ ecb,
                                                    float* __restrict__ out) {
    const int gid = blockIdx.x * 256 + threadIdx.x;   // 16384 = B*N
    const int b = gid & 15;
    const int n = gid >> 4;
    float h[64];
    const unsigned short* l = last + ((size_t)n * 16 + b) * 64;
#pragma unroll
    for (int i = 0; i < 64; ++i) h[i] = bf2f(l[i]);
    for (int o = 0; o < 12; ++o) {
        float acc = ecb[o];
#pragma unroll
        for (int i = 0; i < 64; ++i) acc += h[i] * ecw[o * 64 + i];
        out[((size_t)b * 12 + o) * N_NODES + n] = acc;
    }
}

// ---------------- host orchestration ----------------
extern "C" void kernel_launch(void* const* d_in, const int* in_sizes, int n_in,
                              void* d_out, int out_size, void* d_ws, size_t ws_size,
                              hipStream_t stream) {
    (void)in_sizes; (void)n_in; (void)out_size; (void)ws_size;
    const float* src = (const float*)d_in[0];
    const float* emb = (const float*)d_in[2];
    const float* vew = (const float*)d_in[3];
    const float* gw  = (const float*)d_in[4];
    const float* gb  = (const float*)d_in[5];
    const float* uw  = (const float*)d_in[6];
    const float* ub  = (const float*)d_in[7];
    const float* ecw = (const float*)d_in[8];
    const float* ecb = (const float*)d_in[9];
    float* out = (float*)d_out;

    char* ws = (char*)d_ws;
    auto alloc = [&](size_t bytes) -> char* {
        char* p = ws; ws += (bytes + 255) & ~(size_t)255; return p;
    };
    unsigned short* Aswz = (unsigned short*)alloc((size_t)N_NODES * N_NODES * 2);
    unsigned short* bufA = (unsigned short*)alloc((size_t)P_STEPS * SEQ_T * 2);
    unsigned short* bufB = (unsigned short*)alloc((size_t)P_STEPS * SEQ_T * 2);
    unsigned short* Wg[2], *Wu[2];
    float* bg[2]; float* bu[2];
    for (int l = 0; l < 2; ++l) {
        Wg[l] = (unsigned short*)alloc((size_t)N_NODES * 256 * 128 * 2);
        Wu[l] = (unsigned short*)alloc((size_t)N_NODES * 256 * 64 * 2);
        bg[l] = (float*)alloc((size_t)N_NODES * 128 * 4);
        bu[l] = (float*)alloc((size_t)N_NODES * 64 * 4);
    }
    float* state              = (float*)alloc((size_t)N_NODES * 16 * 64 * 4);
    unsigned short* inp       = (unsigned short*)alloc((size_t)N_NODES * 16 * 128 * 2);
    unsigned short* inp_swz   = (unsigned short*)alloc((size_t)N_NODES * 16 * 128 * 2);
    unsigned short* ax        = (unsigned short*)alloc((size_t)N_NODES * 16 * 128 * 2);
    unsigned short* cand      = (unsigned short*)alloc((size_t)N_NODES * 16 * 128 * 2);
    unsigned short* cand_swz  = (unsigned short*)alloc((size_t)N_NODES * 16 * 128 * 2);
    float* rbuf               = (float*)alloc((size_t)N_NODES * 16 * 64 * 4);

    // precompute
    adj_kernel<<<N_NODES, 128, 0, stream>>>(emb, Aswz);
    patch_kernel<<<(P_STEPS * N_NODES * BATCH) / 256, 256, 0, stream>>>(src, vew, bufA);
    for (int l = 0; l < 2; ++l) {
        wcombine_kernel<<<(N_NODES * 256 * 32) / 256, 256, 0, stream>>>(emb, gw + (size_t)l * 16 * 256 * 128, Wg[l], 128);
        wcombine_kernel<<<(N_NODES * 256 * 16) / 256, 256, 0, stream>>>(emb, uw + (size_t)l * 16 * 256 * 64,  Wu[l], 64);
        bcombine_kernel<<<(N_NODES * 128) / 256, 256, 0, stream>>>(emb, gb + (size_t)l * 16 * 128, bg[l], 128);
        bcombine_kernel<<<(N_NODES * 64) / 256, 256, 0, stream>>>(emb, ub + (size_t)l * 16 * 64,  bu[l], 64);
    }

    // recurrent layers
    for (int l = 0; l < 2; ++l) {
        hipMemsetAsync(state, 0, (size_t)N_NODES * 16 * 64 * 4, stream);
        const unsigned short* seqIn = l ? bufB : bufA;
        unsigned short* seqOut      = l ? bufA : bufB;
        for (int t = 0; t < P_STEPS; ++t) {
            pack_kernel<<<(N_NODES * 16 * 128) / 256, 256, 0, stream>>>(seqIn + (size_t)t * SEQ_T, state, inp, inp_swz);
            gemm_adj_kernel<<<dim3(N_NODES / 16, BATCH), 128, 0, stream>>>(Aswz, inp_swz, ax);
            gate_kernel<<<N_NODES, 128, 0, stream>>>(inp, ax, Wg[l], bg[l], state, cand, cand_swz, rbuf);
            gemm_adj_kernel<<<dim3(N_NODES / 16, BATCH), 128, 0, stream>>>(Aswz, cand_swz, ax);
            update_kernel<<<N_NODES, 128, 0, stream>>>(cand, ax, Wu[l], bu[l], rbuf, state, seqOut + (size_t)t * SEQ_T);
        }
    }
    final_kernel<<<(BATCH * N_NODES) / 256, 256, 0, stream>>>(bufA + (size_t)(P_STEPS - 1) * SEQ_T, ecw, ecb, out);
}
